// CAM_Multimodal_Module_44693429682587
// MI455X (gfx1250) — compile-verified, matrix operand order checked by reading, not run on hardware
//
#include <hip/hip_runtime.h>
#include <hip/hip_bf16.h>

// Problem constants (from reference): B=16, C=512, H=W=48 -> HW=2304
#define BATCH  16
#define CCH    512
#define HWN    2304
#define KC     32      // K-chunk per WMMA pass (one 16x16x32 bf16 WMMA)
#define NKT    (HWN / KC)
#define CBLK   128     // c rows per block (8 waves x 16)
#define DGRP   64      // d columns per accumulator group (4 x 16)
#define DSPLIT 4       // d-range split across blockIdx.z (parallelism)
#define QSTRF  34      // f32 LDS row stride (32 + 2 pad) == TDM pad 2 dwords / 32
#define DSTRF  34
#define DSTRB  40      // bf16 LDS row stride for converted depth tiles

typedef __attribute__((ext_vector_type(8)))  float        v8f;
typedef __attribute__((ext_vector_type(8)))  __bf16       v8bf;
typedef __attribute__((ext_vector_type(16))) __bf16       v16bf;
typedef __attribute__((ext_vector_type(4)))  unsigned int v4u;
typedef __attribute__((ext_vector_type(8)))  int          v8i;
typedef __attribute__((ext_vector_type(4)))  int          v4i;

#if defined(__HIP_DEVICE_COMPILE__) && __has_builtin(__builtin_amdgcn_tensor_load_to_lds)
#define USE_TDM 1
#else
#define USE_TDM 0
#endif

union FragU { v16bf v; v8bf h[2]; };

// B 16-bit fragment per CDNA5 ISA layout (16-bit 16x32):
// lanes 0-15 hold K [0..7] and [16..23]; lanes 16-31 hold K [8..15] and [24..31].
__device__ __forceinline__ v16bf frag16(const __bf16* rowp, int hi) {
  FragU u;
  u.h[0] = *reinterpret_cast<const v8bf*>(rowp + 8 * hi);
  u.h[1] = *reinterpret_cast<const v8bf*>(rowp + 16 + 8 * hi);
  return u.v;
}

__device__ __forceinline__ unsigned int lds_off(const void* p) {
  // LDS aperture generic address: low 32 bits = byte offset into LDS
  return (unsigned int)(uintptr_t)p;
}

#if USE_TDM
// Issue one TDM 2D tile load: tile_dim0 = 32 f32 (128B), tile_dim1 = rows,
// row stride HWN, LDS padding +2 dwords per 32 dwords -> 34-dword LDS rows.
__device__ __forceinline__ void tdm_load_2d(const float* gptr,
                                            unsigned int lds_byte,
                                            unsigned int rows,
                                            unsigned int tensor_rows) {
  const unsigned long long ga = (unsigned long long)(uintptr_t)gptr;
  v4u g0;
  g0[0] = 1u;                                            // count=1 (valid), user mode
  g0[1] = lds_byte;                                      // lds_addr
  g0[2] = (unsigned int)ga;                              // global_addr[31:0]
  g0[3] = (unsigned int)((ga >> 32) & 0x01FFFFFFu)       // global_addr[56:32]
          | (2u << 30);                                  // type=2 ("image")
  v8i g1;
  g1[0] = (int)((2u << 16)        // data_size = 2 -> 4 bytes
              | (1u << 20)        // pad_enable
              | (4u << 22)        // pad_interval: 1<<4 * 8B = 128B = 32 dwords
              | (1u << 25));      // pad_amount: 2 dwords
  g1[1] = (int)((HWN & 0xFFFFu) << 16);                  // tensor_dim0[15:0]
  g1[2] = (int)((HWN >> 16) | ((tensor_rows & 0xFFFFu) << 16)); // dim0 hi, dim1 lo
  g1[3] = (int)((tensor_rows >> 16) | (32u << 16));      // dim1 hi, tile_dim0=32
  g1[4] = (int)rows;                                     // tile_dim1; tile_dim2=0
  g1[5] = (int)HWN;                                      // tensor_dim0_stride[31:0]
  g1[6] = 0;                                             // stride hi, dim1_stride lo
  g1[7] = 0;
  const v4i z4 = {0, 0, 0, 0};
  const v8i z8 = {0, 0, 0, 0, 0, 0, 0, 0};
  // 6-arg form (amdgpu-toolchain clang-23): groups 2/3 + extra group disabled.
  __builtin_amdgcn_tensor_load_to_lds(g0, g1, z4, z4, z8, 0);
}
#endif

__global__ __launch_bounds__(256) void energy_argmax_kernel(
    const float* __restrict__ rgb, const float* __restrict__ depth,
    float* __restrict__ pval, int* __restrict__ pidx)
{
  __shared__ __align__(16) float  qf32[2][CBLK * QSTRF];
  __shared__ __align__(16) float  df32[2][DGRP * DSTRF];
  __shared__ __align__(16) __bf16 dhi[DGRP * DSTRB];
  __shared__ __align__(16) __bf16 dlo[DGRP * DSTRB];

  const int tid  = threadIdx.x;
  const int wave = tid >> 5;
  const int lane = tid & 31;
  const int l16  = lane & 15;
  const int hsel = lane >> 4;
  const int b    = blockIdx.y;
  const int zz   = blockIdx.z;
  const int cblk = blockIdx.x * CBLK;

  const size_t qbase = ((size_t)b * CCH + cblk) * HWN;

  float best_val[8];
  int   best_idx[8];
#pragma unroll
  for (int r = 0; r < 8; ++r) { best_val[r] = -__builtin_inff(); best_idx[r] = 0; }

  const int dg_lo = zz * (CCH / DGRP / DSPLIT);
  const int dg_hi = dg_lo + (CCH / DGRP / DSPLIT);

  for (int dg = dg_lo; dg < dg_hi; ++dg) {
    const size_t dbase = ((size_t)b * CCH + dg * DGRP) * HWN;
    v8f acc[4] = {};

#if USE_TDM
    if (wave == 0) {  // prime the pipeline: kt=0 tiles into buffer 0
      tdm_load_2d(rgb + qbase,   lds_off(&qf32[0][0]), CBLK, CCH);
      tdm_load_2d(depth + dbase, lds_off(&df32[0][0]), DGRP, CCH);
    }
#endif

    for (int kt = 0; kt < NKT; ++kt) {
      const int cur = kt & 1;
      const int nxt = cur ^ 1;

#if USE_TDM
      if (wave == 0) __builtin_amdgcn_s_wait_tensorcnt(0);
      __syncthreads();                         // f32 tiles [cur] ready for all
      if (wave == 0 && kt + 1 < NKT) {         // overlap next DMA with compute
        const int k1 = (kt + 1) * KC;
        tdm_load_2d(rgb + qbase + k1,   lds_off(&qf32[nxt][0]), CBLK, CCH);
        tdm_load_2d(depth + dbase + k1, lds_off(&df32[nxt][0]), DGRP, CCH);
      }
#else
      __syncthreads();
      {  // fallback: cooperative vector loads into the f32 tiles
        const int k0 = kt * KC;
#pragma unroll
        for (int s = 0; s < 4; ++s) {          // Q: 1024 float4
          const int slot = tid + s * 256;
          const int row  = slot >> 3;
          const int c4   = (slot & 7) << 2;
          const float4 v = *reinterpret_cast<const float4*>(
              rgb + qbase + (size_t)row * HWN + k0 + c4);
          float* l = &qf32[cur][row * QSTRF + c4];
          l[0] = v.x; l[1] = v.y; l[2] = v.z; l[3] = v.w;
        }
#pragma unroll
        for (int s = 0; s < 2; ++s) {          // D: 512 float4
          const int slot = tid + s * 256;
          const int row  = slot >> 3;
          const int c4   = (slot & 7) << 2;
          const float4 v = *reinterpret_cast<const float4*>(
              depth + dbase + (size_t)row * HWN + k0 + c4);
          float* l = &df32[cur][row * DSTRF + c4];
          l[0] = v.x; l[1] = v.y; l[2] = v.z; l[3] = v.w;
        }
      }
      __syncthreads();
#endif

      // Cooperative split of the depth tile to bf16 hi/lo (once per block,
      // 2048 elements / 256 threads = 8 each), read stride conflict-free.
#pragma unroll
      for (int s = 0; s < 8; ++s) {
        const int idx = tid + s * 256;
        const int row = idx >> 5;
        const int col = idx & 31;
        const float  x = df32[cur][row * DSTRF + col];
        const __bf16 h = (__bf16)x;
        dhi[row * DSTRB + col] = h;
        dlo[row * DSTRB + col] = (__bf16)(x - (float)h);
      }
      __syncthreads();                         // dhi/dlo ready

      // Per-wave Q fragment: split f32 -> (hi, lo) bf16 directly in registers.
      const float* qrow = &qf32[cur][(wave * 16 + l16) * QSTRF];
      v16bf ah, al;
#pragma unroll
      for (int t = 0; t < 8; ++t) {
        const float  x0 = qrow[8 * hsel + t];
        const float  x1 = qrow[16 + 8 * hsel + t];
        const __bf16 h0 = (__bf16)x0;
        const __bf16 h1 = (__bf16)x1;
        ah[t]     = h0; al[t]     = (__bf16)(x0 - (float)h0);
        ah[8 + t] = h1; al[8 + t] = (__bf16)(x1 - (float)h1);
      }

#if defined(__HIP_DEVICE_COMPILE__)
#pragma unroll
      for (int j = 0; j < 4; ++j) {
        const __bf16* dh = &dhi[(j * 16 + l16) * DSTRB];
        const __bf16* dl = &dlo[(j * 16 + l16) * DSTRB];
        const v16bf bh = frag16(dh, hsel);
        const v16bf bl = frag16(dl, hsel);
        // energy ~= hi*hi + hi*lo + lo*hi  (f32 accumulation)
        acc[j] = __builtin_amdgcn_wmma_f32_16x16x32_bf16(
            false, ah, false, bh, (short)0, acc[j], false, false);
        acc[j] = __builtin_amdgcn_wmma_f32_16x16x32_bf16(
            false, ah, false, bl, (short)0, acc[j], false, false);
        acc[j] = __builtin_amdgcn_wmma_f32_16x16x32_bf16(
            false, al, false, bh, (short)0, acc[j], false, false);
      }
#else
      (void)ah; (void)al;
#endif
    }

    // Fold this d-group into running argmax.
    // C/D layout: VGPR r -> row M = r + 8*(lane/16); lane%16 -> column N.
#pragma unroll
    for (int j = 0; j < 4; ++j) {
      const int d = dg * DGRP + j * 16 + l16;
#pragma unroll
      for (int r = 0; r < 8; ++r) {
        const float v = acc[j][r];
        if (v > best_val[r]) { best_val[r] = v; best_idx[r] = d; }  // strict >: first max wins
      }
    }
  }

  // Reduce across the 16 lanes sharing each C row (xor butterfly stays in-group).
#pragma unroll
  for (int off = 1; off < 16; off <<= 1) {
#pragma unroll
    for (int r = 0; r < 8; ++r) {
      const float ov = __shfl_xor(best_val[r], off, 32);
      const int   oi = __shfl_xor(best_idx[r], off, 32);
      if (ov > best_val[r] || (ov == best_val[r] && oi < best_idx[r])) {
        best_val[r] = ov; best_idx[r] = oi;
      }
    }
  }

  if (l16 == 0) {
#pragma unroll
    for (int r = 0; r < 8; ++r) {
      const int c = cblk + wave * 16 + r + 8 * hsel;
      const size_t o = ((size_t)b * CCH + c) * DSPLIT + zz;
      pval[o] = best_val[r];
      pidx[o] = best_idx[r];
    }
  }
}

// Combine DSPLIT partial argmaxes, then out[b,c,:] = rgb[b,c,:] + depth[b,idx,:]
__global__ __launch_bounds__(256) void gather_add_kernel(
    const float* __restrict__ rgb, const float* __restrict__ depth,
    const float* __restrict__ pval, const int* __restrict__ pidx,
    float* __restrict__ out)
{
  const int bc = blockIdx.x;          // b*CCH + c
  const int b  = bc / CCH;

  float bv = -__builtin_inff();
  int   bi = 0;
#pragma unroll
  for (int z = 0; z < DSPLIT; ++z) {
    const float v = pval[(size_t)bc * DSPLIT + z];
    const int   i = pidx[(size_t)bc * DSPLIT + z];
    if (v > bv || (v == bv && i < bi)) { bv = v; bi = i; }
  }

  const float* rp = rgb   + (size_t)bc * HWN;
  const float* gp = depth + ((size_t)b * CCH + bi) * HWN;
  float*       op = out   + (size_t)bc * HWN;
  for (int n = threadIdx.x; n < HWN; n += 256)
    op[n] = rp[n] + gp[n];
}

extern "C" void kernel_launch(void* const* d_in, const int* in_sizes, int n_in,
                              void* d_out, int out_size, void* d_ws, size_t ws_size,
                              hipStream_t stream) {
  (void)in_sizes; (void)n_in; (void)out_size; (void)ws_size;
  const float* rgb   = (const float*)d_in[0];
  const float* depth = (const float*)d_in[1];
  float* out = (float*)d_out;

  float* pval = (float*)d_ws;
  int*   pidx = (int*)((char*)d_ws + sizeof(float) * (size_t)BATCH * CCH * DSPLIT);

  dim3 grid1(CCH / CBLK, BATCH, DSPLIT);
  hipLaunchKernelGGL(energy_argmax_kernel, grid1, dim3(256), 0, stream,
                     rgb, depth, pval, pidx);
  hipLaunchKernelGGL(gather_add_kernel, dim3(BATCH * CCH), dim3(256), 0, stream,
                     rgb, depth, pval, pidx, out);
}
